// TritonConvAttention_41266045780917
// MI455X (gfx1250) — matrix operand order repaired and checked by reading, not compiled, not run on hardware
//
#include <hip/hip_runtime.h>
#include <math.h>

// Fused conv-attention for MI455X (gfx1250, wave32, WMMA).
// One workgroup = (b, h, 32-query block). Per 32-key step:
//   1. stage K (f32->bf16) and V (f32->bf16, TRANSPOSED) into LDS;
//      prefetch next block's K/V into cache (global_prefetch_b8)
//   2. QK^T via v_wmma_f32_16x16x32_bf16 over a 48x64 halo tile (causal+pad masked)
//   3. branch-free 6x11 sliding-window conv on the score tile (register segments)
//   4. online softmax (shfl row reductions), P -> bf16 LDS (packed uint4 stores)
//   5. PV via v_wmma_f32_16x16x32_bf16; all fragment LDS reads are contiguous
//      b128s from conflict-free padded strides.

#define NB   2
#define NH   12
#define SEQ  1024
#define DHD  128
#define KQ_  6
#define KK_  11
#define BQ   32           // query rows per workgroup
#define QR   48           // staged score rows (16-row top halo)
#define BK   32           // keys per step
#define KC   64           // staged score cols (16-col halo each side)
#define SSTR 65           // padded score row stride (floats) -> conflict-free
#define QSTR 136          // Qs/Ks row stride in u16 (272B = 17*16B): aligned + conflict-free
#define KSTR 136
#define VSTR 40           // transposed-V row stride in u16 (80B = 5*16B)
#define PSTR 40           // P row stride in u16
#define NTH  128          // 4 waves

typedef __attribute__((ext_vector_type(16))) __bf16 v16bf;
typedef __attribute__((ext_vector_type(2)))  __bf16 v2bf;
typedef __attribute__((ext_vector_type(8)))  float  v8f;

union BF16Frag { v16bf v; unsigned short s[16]; };
union BF16Pair { v2bf v; unsigned int u; };

__device__ __forceinline__ unsigned short f2bf(float f) {
  unsigned int u = __float_as_uint(f);
  u += 0x7FFFu + ((u >> 16) & 1u);          // round-to-nearest-even
  return (unsigned short)(u >> 16);
}

// Pack two f32 into one dword of bf16 (single v_cvt_pk_bf16_f32 when available).
__device__ __forceinline__ unsigned int pack2bf(float a, float b) {
#if __has_builtin(__builtin_amdgcn_cvt_pk_bf16_f32)
  BF16Pair p;
  p.v = __builtin_amdgcn_cvt_pk_bf16_f32(a, b);
  return p.u;
#else
  return (unsigned)f2bf(a) | ((unsigned)f2bf(b) << 16);
#endif
}

__global__ __launch_bounds__(NTH)
void conv_attn_fused(const float* __restrict__ Qg, const float* __restrict__ Kg,
                     const float* __restrict__ Vg, const float* __restrict__ Wg,
                     float* __restrict__ Og) {
  __shared__ unsigned short Qs[QR * QSTR];     // ~13 KB bf16 Q tile (48 x 128)
  __shared__ unsigned short Ks[KC * KSTR];     // ~17 KB bf16 K tile (64 x 128)
  __shared__ unsigned short Vt[DHD * VSTR];    // ~10 KB bf16 V^T tile (128 x 32)
  __shared__ float          Sraw[QR * SSTR];   // ~12 KB raw masked scores
  __shared__ unsigned short Pl[BQ * PSTR];     // ~2.5 KB bf16 softmax P
  __shared__ float          Wl[KQ_ * KK_];
  __shared__ float          alphaL[BQ];
  __shared__ float          lsumL[BQ];

  const int q0   = blockIdx.x * BQ;
  const int h    = blockIdx.y;
  const int b    = blockIdx.z;
  const int tid  = threadIdx.x;
  const int lane = tid & 31;
  const int wave = tid >> 5;
  const int qbase = q0 - 16;

  const long long bh = (long long)(b * NH + h);
  const float* Qbh = Qg + bh * SEQ * DHD;
  const float* Kbh = Kg + bh * SEQ * DHD;
  const float* Vbh = Vg + bh * SEQ * DHD;
  float*       Obh = Og + bh * SEQ * DHD;

  // ---- stage conv weights + Q tile (once per workgroup) ----
  if (tid < KQ_ * KK_) Wl[tid] = Wg[h * KQ_ * KK_ + tid];
  for (int idx = tid; idx < QR * (DHD / 4); idx += NTH) {
    int qr = idx >> 5, d4 = idx & 31;
    int q = qbase + qr;
    float4 v = make_float4(0.f, 0.f, 0.f, 0.f);
    if (q >= 0) v = *(const float4*)(Qbh + (long long)q * DHD + d4 * 4);
    uint2 p;
    p.x = pack2bf(v.x, v.y);
    p.y = pack2bf(v.z, v.w);
    *(uint2*)&Qs[qr * QSTR + d4 * 4] = p;     // 272B stride keeps 8B alignment
  }

  v8f oacc[4];
  #pragma unroll
  for (int t = 0; t < 4; ++t)
    #pragma unroll
    for (int i = 0; i < 8; ++i) oacc[t][i] = 0.f;

  float m_run = -INFINITY, l_run = 0.f;
  const int r  = tid >> 2;          // softmax row owned by this thread
  const int c0 = (tid & 3) * 8;     // 8-col chunk within the 32-key block

  const int nkb = q0 / BK + 1;      // causal: key blocks 0 .. q0/32
  for (int kb = 0; kb < nkb; ++kb) {
    const int k0 = kb * BK;
    const int kbase = k0 - 16;
    __syncthreads();                // prior PV / conv readers done

    // ---- stage K halo tile (row-major bf16) ----
    for (int idx = tid; idx < KC * (DHD / 4); idx += NTH) {
      int kr = idx >> 5, d4 = idx & 31;
      int key = kbase + kr;
      float4 v = make_float4(0.f, 0.f, 0.f, 0.f);
      if (key >= 0 && key < SEQ) v = *(const float4*)(Kbh + (long long)key * DHD + d4 * 4);
      uint2 p;
      p.x = pack2bf(v.x, v.y);
      p.y = pack2bf(v.z, v.w);
      *(uint2*)&Ks[kr * KSTR + d4 * 4] = p;
    }
    // ---- stage V TRANSPOSED: Vt[d][key] (coalesced reads: lane==d) ----
    {
      int d = tid;                               // 0..127
      #pragma unroll
      for (int kg = 0; kg < BK / 4; ++kg) {
        float v0 = Vbh[(long long)(k0 + kg * 4 + 0) * DHD + d];
        float v1 = Vbh[(long long)(k0 + kg * 4 + 1) * DHD + d];
        float v2 = Vbh[(long long)(k0 + kg * 4 + 2) * DHD + d];
        float v3 = Vbh[(long long)(k0 + kg * 4 + 3) * DHD + d];
        uint2 p;
        p.x = pack2bf(v0, v1);
        p.y = pack2bf(v2, v3);
        *(uint2*)&Vt[d * VSTR + kg * 4] = p;
      }
    }
    // ---- prefetch next block's K/V lines into cache (global_prefetch_b8) ----
    if (kb + 1 < nkb) {
      const int nk0 = k0 + BK;
      for (int t = tid; t < KC * 4; t += NTH) {       // 64 K rows x 4 cachelines
        int row = t >> 2, ln = t & 3;
        int key = nk0 - 16 + row;
        if (key >= 0 && key < SEQ)
          __builtin_prefetch(Kbh + (long long)key * DHD + ln * 32, 0, 1);
      }
      {
        int row = tid >> 2, ln = tid & 3;             // 32 V rows x 4 cachelines
        __builtin_prefetch(Vbh + (long long)(nk0 + row) * DHD + ln * 32, 0, 1);
      }
    }
    __syncthreads();

    // ---- QK^T: wave owns 16-key column strip, 3 row tiles, K=128 in 4 steps ----
    v8f sacc[3];
    #pragma unroll
    for (int rt = 0; rt < 3; ++rt)
      #pragma unroll
      for (int i = 0; i < 8; ++i) sacc[rt][i] = 0.f;

    #pragma unroll
    for (int dc = 0; dc < 4; ++dc) {
      const int dOff = dc * 32;
      BF16Frag bf;                                   // B = K^T chunk (32d x 16keys)
      {
        int keyL = wave * 16 + (lane & 15);
        int kst  = (lane < 16) ? 0 : 16;
        const unsigned short* p = &Ks[keyL * KSTR + dOff + kst];
        #pragma unroll
        for (int j = 0; j < 16; ++j) bf.s[j] = p[j];   // 2x ds_load_b128
      }
      #pragma unroll
      for (int rt = 0; rt < 3; ++rt) {
        BF16Frag af;                                 // A = Q rows (16q x 32d)
        int row = rt * 16 + (lane & 15);
        int kh  = (lane < 16) ? 0 : 16;
        const unsigned short* p = &Qs[row * QSTR + dOff + kh];
        #pragma unroll
        for (int j = 0; j < 16; ++j) af.s[j] = p[j];
        sacc[rt] = __builtin_amdgcn_wmma_f32_16x16x32_bf16(
            false, af.v, false, bf.v, (short)0, sacc[rt], false, false);
      }
    }

    // ---- store causally masked raw scores (zeros outside, as the reference) ----
    #pragma unroll
    for (int rt = 0; rt < 3; ++rt)
      #pragma unroll
      for (int i = 0; i < 8; ++i) {
        int row  = rt * 16 + i + ((lane >> 4) << 3);
        int colL = wave * 16 + (lane & 15);
        int q    = qbase + row;
        int key  = kbase + colL;
        bool ok  = (q >= 0) && (key >= 0) && (key < SEQ) && (key <= q);
        Sraw[row * SSTR + colL] = ok ? sacc[rt][i] : 0.f;
      }
    __syncthreads();

    // ---- branch-free 6x11 conv: sliding register window over zero-masked
    //      scores; causal -1e6 fill applied afterwards with a select ----
    const int q = q0 + r;
    float logit[8];
    #pragma unroll
    for (int j = 0; j < 8; ++j) logit[j] = 0.f;
    {
      const float* sbase = &Sraw[(r + 11) * SSTR + (c0 + 11)];
      #pragma unroll 1
      for (int a = 0; a < KQ_; ++a) {
        float seg[KK_ + 7];                     // 18-wide window for 8 outputs
        #pragma unroll
        for (int t2 = 0; t2 < KK_ + 7; ++t2) seg[t2] = sbase[t2];
        const float* wrow = &Wl[a * KK_];
        #pragma unroll
        for (int t2 = 0; t2 < KK_; ++t2) {
          float w = wrow[t2];
          #pragma unroll
          for (int j = 0; j < 8; ++j) logit[j] = fmaf(w, seg[t2 + j], logit[j]);
        }
        sbase += SSTR;
      }
    }
    #pragma unroll
    for (int j = 0; j < 8; ++j) {
      int key = k0 + c0 + j;
      logit[j] = (key > q) ? -1.0e6f : logit[j] * 0.08838834764831845f;
    }

    float mb = logit[0];
    #pragma unroll
    for (int j = 1; j < 8; ++j) mb = fmaxf(mb, logit[j]);
    mb = fmaxf(mb, __shfl_xor(mb, 1, 32));
    mb = fmaxf(mb, __shfl_xor(mb, 2, 32));
    float m_new = fmaxf(m_run, mb);
    float alpha = __expf(m_run - m_new);
    float ls = 0.f;
    unsigned int pw[4];
    #pragma unroll
    for (int j2 = 0; j2 < 4; ++j2) {
      float pa = __expf(logit[2 * j2]     - m_new);
      float pb = __expf(logit[2 * j2 + 1] - m_new);
      ls += pa + pb;
      pw[j2] = pack2bf(pa, pb);
    }
    {
      uint4 pv4; pv4.x = pw[0]; pv4.y = pw[1]; pv4.z = pw[2]; pv4.w = pw[3];
      *(uint4*)&Pl[r * PSTR + c0] = pv4;             // 16B-aligned packed store
    }
    ls += __shfl_xor(ls, 1, 32);
    ls += __shfl_xor(ls, 2, 32);
    l_run = l_run * alpha + ls;
    m_run = m_new;
    if ((tid & 3) == 0) alphaL[r] = alpha;
    __syncthreads();

    // ---- rescale O accumulators, PV: wave owns 32 D-columns ----
    BF16Frag ap[2];
    #pragma unroll
    for (int rt = 0; rt < 2; ++rt) {
      int row = rt * 16 + (lane & 15);
      int kh  = (lane < 16) ? 0 : 16;
      const unsigned short* p = &Pl[row * PSTR + kh];
      #pragma unroll
      for (int j = 0; j < 16; ++j) ap[rt].s[j] = p[j];
    }
    BF16Frag bv[2];
    #pragma unroll
    for (int ct = 0; ct < 2; ++ct) {
      int col = wave * 32 + ct * 16 + (lane & 15);   // D column
      int kst = (lane < 16) ? 0 : 16;                // key half
      const unsigned short* p = &Vt[col * VSTR + kst];
      #pragma unroll
      for (int j = 0; j < 16; ++j) bv[ct].s[j] = p[j];  // 2x ds_load_b128
    }
    #pragma unroll
    for (int t = 0; t < 4; ++t) {
      int rt = t >> 1, ct = t & 1;
      v8f o = oacc[t];
      #pragma unroll
      for (int i = 0; i < 8; ++i) {
        int row = rt * 16 + i + ((lane >> 4) << 3);
        o[i] *= alphaL[row];
      }
      oacc[t] = __builtin_amdgcn_wmma_f32_16x16x32_bf16(
          false, ap[rt].v, false, bv[ct].v, (short)0, o, false, false);
    }
  }

  if ((tid & 3) == 0) lsumL[r] = l_run;
  __syncthreads();

  // ---- normalize + write O ----
  #pragma unroll
  for (int t = 0; t < 4; ++t) {
    int rt = t >> 1, ct = t & 1;
    #pragma unroll
    for (int i = 0; i < 8; ++i) {
      int row = rt * 16 + i + ((lane >> 4) << 3);
      int col = wave * 32 + ct * 16 + (lane & 15);
      int q   = q0 + row;
      Obh[(long long)q * DHD + col] = oacc[t][i] / lsumL[row];
    }
  }
}

extern "C" void kernel_launch(void* const* d_in, const int* in_sizes, int n_in,
                              void* d_out, int out_size, void* d_ws, size_t ws_size,
                              hipStream_t stream) {
  (void)in_sizes; (void)n_in; (void)out_size; (void)d_ws; (void)ws_size;
  const float* Q = (const float*)d_in[0];
  const float* K = (const float*)d_in[1];
  const float* V = (const float*)d_in[2];
  const float* W = (const float*)d_in[3];
  float* O = (float*)d_out;
  dim3 grid(SEQ / BQ, NH, NB);   // 32 x 12 x 2 = 768 workgroups
  dim3 block(NTH);
  hipLaunchKernelGGL(conv_attn_fused, grid, block, 0, stream, Q, K, V, W, O);
}
